// Multi_CrossAttention_36215164240131
// MI455X (gfx1250) — matrix operand order repaired and checked by reading, not compiled, not run on hardware
//
#include <hip/hip_runtime.h>
#include <stdint.h>

// Multi-head cross attention for MI455X (gfx1250), wave32 + WMMA bf16.
//   B=4, S=2048, D=1024, H=16, HD=64
// Pipeline: qkv projection (WMMA GEMM, f32->bf16) -> flash attention (WMMA,
// online softmax, key-padding mask, async double-buffered K/V staging) ->
// output projection (WMMA GEMM + bias, async A staging).
// Workspace layout (needs 64 MiB): qb | kb | vb | ob, 16 MiB bf16 each.

#define BQ   4
#define SEQ  2048
#define DIM  1024
#define NH   16
#define HD   64
#define ROWS (BQ * SEQ)          // 8192
#define NEGV  (-1000000000.0f)
#define SCALE 0.125f             // 1/sqrt(64)

typedef __bf16 v16bf __attribute__((ext_vector_type(16)));
typedef __bf16 v8bf  __attribute__((ext_vector_type(8)));
typedef float  v8f   __attribute__((ext_vector_type(8)));

union FragBF { v16bf v; v8bf h[2]; };

static __device__ __forceinline__ v8f wmma_bf16(const FragBF& a, const FragBF& b, v8f c) {
  // D = A(16x32 bf16) * B(32x16 bf16) + C(16x16 f32)
  return __builtin_amdgcn_wmma_f32_16x16x32_bf16(false, a.v, false, b.v,
                                                 (short)0, c, false, false);
}

// Async 16B copy global -> LDS (ASYNCcnt). Low 32 bits of a generic pointer
// to __shared__ are the LDS byte address (ISA 10.2 aperture rule).
static __device__ __forceinline__ void async_b128(const __bf16* g, __bf16* l) {
  const unsigned long long ga = (unsigned long long)(uintptr_t)g;
  const unsigned int       la = (unsigned int)(uintptr_t)l;
  asm volatile("global_load_async_to_lds_b128 %0, %1, off"
               :: "v"(la), "v"(ga) : "memory");
}
#define WAIT_ASYNC() asm volatile("s_wait_asynccnt 0" ::: "memory")

#define APAD 40   // LDS row stride (elements) for 32-wide K tiles (+8 pad)
#define BPAD 40
#define KP   72   // K_s row stride: 64 hd + 8 pad
#define VP   40   // V_s row stride: 32 keys + 8 pad
#define PP   40   // P_s row stride: 32 keys + 8 pad

// ---------------------------------------------------------------------------
// Kernel 1: Q/K/V projections. grid = (ROWS/128, DIM/64, 3).
// which==0: q = x@Wq -> [B,H,S,HD]; which==1: k = y@Wk -> [B,H,S,HD];
// which==2: v = y@Wv -> [B,H,HD,S] (transposed for the PV B-fragment reads).
// ---------------------------------------------------------------------------
__global__ __launch_bounds__(256)
void proj_qkv_kernel(const float* __restrict__ x, const float* __restrict__ y,
                     const float* __restrict__ Wq, const float* __restrict__ Wk,
                     const float* __restrict__ Wv,
                     __bf16* __restrict__ qb, __bf16* __restrict__ kb,
                     __bf16* __restrict__ vb)
{
  __shared__ __bf16 As[128 * APAD];  // A tile [row][k], bf16
  __shared__ __bf16 Bs[64 * BPAD];   // W tile transposed: [col][k], bf16

  const int which = blockIdx.z;
  const float* src = (which == 0) ? x : y;
  const float* W   = (which == 0) ? Wq : (which == 1) ? Wk : Wv;

  const int mbase = blockIdx.x * 128;
  const int nbase = blockIdx.y * 64;
  const int tid   = threadIdx.x;
  const int lane  = tid & 31;
  const int wav   = tid >> 5;        // 0..7
  const int wm    = wav >> 1;        // 0..3 (M)
  const int wn    = wav & 1;         // 0..1 (N)
  const int l15   = lane & 15;
  const int lhalf = lane >> 4;

  v8f acc[2][2] = {};

  for (int kk = 0; kk < DIM; kk += 32) {
    // Stage A: 128 rows x 32 cols f32 -> bf16. 2 threads per row.
    {
      const int r = tid >> 1, half = tid & 1;
      const float4* p = (const float4*)(src + (size_t)(mbase + r) * DIM + kk + half * 16);
      float4 f0 = p[0], f1 = p[1], f2 = p[2], f3 = p[3];
      v8bf c0 = { (__bf16)f0.x,(__bf16)f0.y,(__bf16)f0.z,(__bf16)f0.w,
                  (__bf16)f1.x,(__bf16)f1.y,(__bf16)f1.z,(__bf16)f1.w };
      v8bf c1 = { (__bf16)f2.x,(__bf16)f2.y,(__bf16)f2.z,(__bf16)f2.w,
                  (__bf16)f3.x,(__bf16)f3.y,(__bf16)f3.z,(__bf16)f3.w };
      *(v8bf*)&As[r * APAD + half * 16]     = c0;
      *(v8bf*)&As[r * APAD + half * 16 + 8] = c1;
    }
    // Stage B transposed: W[kk..kk+31][nbase..nbase+63] -> Bs[col][k].
    {
      const int krow = tid >> 3, cg = tid & 7;
      const float4* p = (const float4*)(W + (size_t)(kk + krow) * DIM + nbase + cg * 8);
      float4 f0 = p[0], f1 = p[1];
      const float fv[8] = { f0.x,f0.y,f0.z,f0.w, f1.x,f1.y,f1.z,f1.w };
      #pragma unroll
      for (int j = 0; j < 8; ++j)
        Bs[(cg * 8 + j) * BPAD + krow] = (__bf16)fv[j];
    }
    __syncthreads();

    FragBF a[2], bfr[2];
    const int akrun = lhalf * 8;   // A-frag: K runs at 8*(l>>4) and +16
    #pragma unroll
    for (int mt = 0; mt < 2; ++mt) {
      const int row = wm * 32 + mt * 16 + l15;
      a[mt].h[0] = *(const v8bf*)&As[row * APAD + akrun];
      a[mt].h[1] = *(const v8bf*)&As[row * APAD + akrun + 16];
    }
    const int bkrun = lhalf * 8;   // B-frag: K runs at 8*(l>>4) and +16
    #pragma unroll
    for (int nt = 0; nt < 2; ++nt) {
      const int col = wn * 32 + nt * 16 + l15;
      bfr[nt].h[0] = *(const v8bf*)&Bs[col * BPAD + bkrun];
      bfr[nt].h[1] = *(const v8bf*)&Bs[col * BPAD + bkrun + 16];
    }
    #pragma unroll
    for (int mt = 0; mt < 2; ++mt)
      #pragma unroll
      for (int nt = 0; nt < 2; ++nt)
        acc[mt][nt] = wmma_bf16(a[mt], bfr[nt], acc[mt][nt]);
    __syncthreads();
  }

  __bf16* dst = (which == 0) ? qb : (which == 1) ? kb : vb;
  #pragma unroll
  for (int mt = 0; mt < 2; ++mt) {
    #pragma unroll
    for (int nt = 0; nt < 2; ++nt) {
      #pragma unroll
      for (int r = 0; r < 8; ++r) {
        const int grow = mbase + wm * 32 + mt * 16 + r + 8 * lhalf;  // C/D row
        const int gcol = nbase + wn * 32 + nt * 16 + l15;            // C/D col
        const int bb = grow >> 11, ss = grow & 2047;
        const int hh = gcol >> 6,  hd = gcol & 63;
        size_t idx;
        if (which < 2) idx = ((size_t)(bb * NH + hh) * SEQ + ss) * HD + hd;
        else           idx = ((size_t)(bb * NH + hh) * HD + hd) * SEQ + ss;
        dst[idx] = (__bf16)acc[mt][nt][r];
      }
    }
  }
}

// ---------------------------------------------------------------------------
// Kernel 2: flash attention. grid = (SEQ/128, B*H). 8 waves/block, one
// 16-query tile per wave; K/V tiles (32 keys) double-buffered in LDS via
// GLOBAL_LOAD_ASYNC_TO_LDS_B128 so global latency overlaps the WMMAs.
// ---------------------------------------------------------------------------
__global__ __launch_bounds__(256)
void flash_attn_kernel(const __bf16* __restrict__ qb, const __bf16* __restrict__ kb,
                       const __bf16* __restrict__ vb, const int* __restrict__ ls,
                       const int* __restrict__ pad_ptr, __bf16* __restrict__ ob)
{
  __shared__ __bf16 Ks[2][32 * KP];     // [buf][key][hd]
  __shared__ __bf16 Vs[2][64 * VP];     // [buf][hd][key]
  __shared__ __bf16 Ps[8 * 16 * PP];    // per-wave probs [row][key]
  __shared__ int    Ms[2][32];          // key-padding mask per buffer

  const int bh   = blockIdx.y;          // b*H + h
  const int bq   = bh >> 4;             // batch
  const int tid  = threadIdx.x;
  const int lane = tid & 31;
  const int wav  = tid >> 5;
  const int q0   = blockIdx.x * 128 + wav * 16;
  const int pad  = *pad_ptr;
  const int l15  = lane & 15;
  const int lhalf = lane >> 4;

  // Q fragments held for the whole key loop: 2 chunks over HD=64.
  FragBF qf[2];
  {
    const __bf16* qrow = qb + ((size_t)bh * SEQ + q0 + l15) * HD;
    #pragma unroll
    for (int c = 0; c < 2; ++c) {
      qf[c].h[0] = *(const v8bf*)(qrow + c * 32 + lhalf * 8);
      qf[c].h[1] = *(const v8bf*)(qrow + c * 32 + lhalf * 8 + 16);
    }
  }

  // Async staging of one 32-key K/V tile into buffer `buf`.
  auto stage_tile = [&](int buf, int kt) {
    const int key = tid >> 3, hg = tid & 7;
    async_b128(kb + ((size_t)bh * SEQ + kt + key) * HD + hg * 8,
               &Ks[buf][key * KP + hg * 8]);
    const int hd = tid >> 2, kg = tid & 3;
    async_b128(vb + ((size_t)bh * HD + hd) * SEQ + kt + kg * 8,
               &Vs[buf][hd * VP + kg * 8]);
    if (tid < 32) Ms[buf][tid] = (ls[bq * SEQ + kt + tid] == pad) ? 1 : 0;
  };

  float m_run[8], l_run[8];
  v8f accO[4] = {};
  #pragma unroll
  for (int r = 0; r < 8; ++r) { m_run[r] = -3.0e38f; l_run[r] = 0.0f; }

  __bf16* ps = Ps + wav * 16 * PP;

  stage_tile(0, 0);
  WAIT_ASYNC();
  __syncthreads();

  for (int kt = 0; kt < SEQ; kt += 32) {
    const int cur = (kt >> 5) & 1;
    if (kt + 32 < SEQ) stage_tile(cur ^ 1, kt + 32);  // prefetch next tile

    const __bf16* Kc = Ks[cur];
    const __bf16* Vc = Vs[cur];

    // Scores: two 16x16 tiles, each accumulated over HD=64 (2 WMMAs).
    v8f s0 = {}, s1 = {};
    #pragma unroll
    for (int c = 0; c < 2; ++c) {
      FragBF kf0, kf1;
      const int off = c * 32 + lhalf * 8;
      kf0.h[0] = *(const v8bf*)&Kc[l15 * KP + off];
      kf0.h[1] = *(const v8bf*)&Kc[l15 * KP + off + 16];
      kf1.h[0] = *(const v8bf*)&Kc[(16 + l15) * KP + off];
      kf1.h[1] = *(const v8bf*)&Kc[(16 + l15) * KP + off + 16];
      s0 = wmma_bf16(qf[c], kf0, s0);
      s1 = wmma_bf16(qf[c], kf1, s1);
    }

    // Mask BEFORE scaling (faithful to reference masked_fill), then online
    // softmax. C/D layout: vgpr r holds row r+8*(l>>4), col l&15 — so a row
    // reduction is a butterfly across the 16-lane column group.
    const bool mk0 = Ms[cur][l15] != 0;
    const bool mk1 = Ms[cur][16 + l15] != 0;

    float p0[8], p1[8], alpha[8];
    #pragma unroll
    for (int r = 0; r < 8; ++r) {
      p0[r] = (mk0 ? NEGV : s0[r]) * SCALE;
      p1[r] = (mk1 ? NEGV : s1[r]) * SCALE;
    }
    #pragma unroll
    for (int r = 0; r < 8; ++r) {
      float t = fmaxf(p0[r], p1[r]);
      t = fmaxf(t, __shfl_xor(t, 1));
      t = fmaxf(t, __shfl_xor(t, 2));
      t = fmaxf(t, __shfl_xor(t, 4));
      t = fmaxf(t, __shfl_xor(t, 8));
      const float mnew = fmaxf(m_run[r], t);
      alpha[r] = __expf(m_run[r] - mnew);
      m_run[r] = mnew;
      p0[r] = __expf(p0[r] - mnew);
      p1[r] = __expf(p1[r] - mnew);
      float rs = p0[r] + p1[r];
      rs += __shfl_xor(rs, 1);
      rs += __shfl_xor(rs, 2);
      rs += __shfl_xor(rs, 4);
      rs += __shfl_xor(rs, 8);
      l_run[r] = l_run[r] * alpha[r] + rs;
    }
    #pragma unroll
    for (int ht = 0; ht < 4; ++ht)
      #pragma unroll
      for (int r = 0; r < 8; ++r)
        accO[ht][r] = accO[ht][r] * alpha[r];

    // Spill probs to per-wave LDS (bf16) and re-read in A-fragment layout.
    #pragma unroll
    for (int r = 0; r < 8; ++r) {
      const int row = r + 8 * lhalf;
      ps[row * PP + l15]      = (__bf16)p0[r];
      ps[row * PP + 16 + l15] = (__bf16)p1[r];
    }
    __asm__ volatile("s_wait_dscnt 0" ::: "memory");  // wave-internal LDS RAW

    FragBF pf;
    pf.h[0] = *(const v8bf*)&ps[l15 * PP + lhalf * 8];
    pf.h[1] = *(const v8bf*)&ps[l15 * PP + lhalf * 8 + 16];
    #pragma unroll
    for (int ht = 0; ht < 4; ++ht) {
      FragBF vf;
      vf.h[0] = *(const v8bf*)&Vc[(ht * 16 + l15) * VP + lhalf * 8];
      vf.h[1] = *(const v8bf*)&Vc[(ht * 16 + l15) * VP + lhalf * 8 + 16];
      accO[ht] = wmma_bf16(pf, vf, accO[ht]);
    }

    if (kt + 32 < SEQ) WAIT_ASYNC();  // next tile landed in the other buffer
    __syncthreads();
  }

  // Normalize and store O head-split bf16 [B,H,S,HD].
  #pragma unroll
  for (int ht = 0; ht < 4; ++ht) {
    #pragma unroll
    for (int r = 0; r < 8; ++r) {
      const int row = q0 + r + 8 * lhalf;
      const int hd  = ht * 16 + l15;
      ob[((size_t)bh * SEQ + row) * HD + hd] = (__bf16)(accO[ht][r] / l_run[r]);
    }
  }
}

// ---------------------------------------------------------------------------
// Kernel 3: out = O(head-split bf16) @ Wo + bo, f32 output. grid=(64,16).
// A tile staged with async global->LDS copies (already bf16).
// ---------------------------------------------------------------------------
__global__ __launch_bounds__(256)
void out_proj_kernel(const __bf16* __restrict__ ob, const float* __restrict__ Wo,
                     const float* __restrict__ bo, float* __restrict__ out)
{
  __shared__ __bf16 As[128 * APAD];
  __shared__ __bf16 Bs[64 * BPAD];

  const int mbase = blockIdx.x * 128;
  const int nbase = blockIdx.y * 64;
  const int tid   = threadIdx.x;
  const int lane  = tid & 31;
  const int wav   = tid >> 5;
  const int wm    = wav >> 1, wn = wav & 1;
  const int l15   = lane & 15;
  const int lhalf = lane >> 4;
  const int bb    = mbase >> 11;      // whole block within one batch
  const int sbase = mbase & 2047;

  v8f acc[2][2] = {};

  for (int kk = 0; kk < DIM; kk += 32) {
    const int hh = kk >> 6, hdb = kk & 63;  // K chunk lies within one head
    {
      const int r = tid >> 1, half = tid & 1;
      const __bf16* p = ob + ((size_t)(bb * NH + hh) * SEQ + sbase + r) * HD + hdb + half * 16;
      async_b128(p,     &As[r * APAD + half * 16]);
      async_b128(p + 8, &As[r * APAD + half * 16 + 8]);
    }
    {
      const int krow = tid >> 3, cg = tid & 7;
      const float4* p = (const float4*)(Wo + (size_t)(kk + krow) * DIM + nbase + cg * 8);
      float4 f0 = p[0], f1 = p[1];
      const float fv[8] = { f0.x,f0.y,f0.z,f0.w, f1.x,f1.y,f1.z,f1.w };
      #pragma unroll
      for (int j = 0; j < 8; ++j)
        Bs[(cg * 8 + j) * BPAD + krow] = (__bf16)fv[j];
    }
    WAIT_ASYNC();
    __syncthreads();

    FragBF a[2], bfr[2];
    const int akrun = lhalf * 8;
    #pragma unroll
    for (int mt = 0; mt < 2; ++mt) {
      const int row = wm * 32 + mt * 16 + l15;
      a[mt].h[0] = *(const v8bf*)&As[row * APAD + akrun];
      a[mt].h[1] = *(const v8bf*)&As[row * APAD + akrun + 16];
    }
    const int bkrun = lhalf * 8;
    #pragma unroll
    for (int nt = 0; nt < 2; ++nt) {
      const int col = wn * 32 + nt * 16 + l15;
      bfr[nt].h[0] = *(const v8bf*)&Bs[col * BPAD + bkrun];
      bfr[nt].h[1] = *(const v8bf*)&Bs[col * BPAD + bkrun + 16];
    }
    #pragma unroll
    for (int mt = 0; mt < 2; ++mt)
      #pragma unroll
      for (int nt = 0; nt < 2; ++nt)
        acc[mt][nt] = wmma_bf16(a[mt], bfr[nt], acc[mt][nt]);
    __syncthreads();
  }

  #pragma unroll
  for (int mt = 0; mt < 2; ++mt) {
    #pragma unroll
    for (int nt = 0; nt < 2; ++nt) {
      #pragma unroll
      for (int r = 0; r < 8; ++r) {
        const int grow = mbase + wm * 32 + mt * 16 + r + 8 * lhalf;
        const int gcol = nbase + wn * 32 + nt * 16 + l15;
        out[(size_t)grow * DIM + gcol] = acc[mt][nt][r] + bo[gcol];
      }
    }
  }
}

// ---------------------------------------------------------------------------
extern "C" void kernel_launch(void* const* d_in, const int* in_sizes, int n_in,
                              void* d_out, int out_size, void* d_ws, size_t ws_size,
                              hipStream_t stream) {
  (void)in_sizes; (void)n_in; (void)out_size; (void)ws_size;
  const float* x   = (const float*)d_in[0];
  const float* y   = (const float*)d_in[1];
  const int*   ls  = (const int*)d_in[2];   // log_seqs (int)
  const int*   pad = (const int*)d_in[3];   // pad_id scalar
  const float* Wq  = (const float*)d_in[4];
  const float* Wk  = (const float*)d_in[5];
  const float* Wv  = (const float*)d_in[6];
  const float* Wo  = (const float*)d_in[7];
  const float* bo  = (const float*)d_in[8];
  float* out = (float*)d_out;

  char* ws = (char*)d_ws;
  const size_t matBytes = (size_t)ROWS * DIM * sizeof(__bf16);  // 16 MiB each
  __bf16* qb  = (__bf16*)(ws);
  __bf16* kb  = (__bf16*)(ws + matBytes);
  __bf16* vb  = (__bf16*)(ws + 2 * matBytes);
  __bf16* obf = (__bf16*)(ws + 3 * matBytes);

  proj_qkv_kernel<<<dim3(ROWS / 128, DIM / 64, 3), 256, 0, stream>>>(
      x, y, Wq, Wk, Wv, qb, kb, vb);
  flash_attn_kernel<<<dim3(SEQ / 128, BQ * NH), 256, 0, stream>>>(
      qb, kb, vb, ls, pad, obf);
  out_proj_kernel<<<dim3(ROWS / 128, DIM / 64), 256, 0, stream>>>(
      obf, Wo, bo, out);
}
